// StitchDecoder_81389630259657
// MI455X (gfx1250) — matrix operand order repaired and checked by reading, not compiled, not run on hardware
//
#include <hip/hip_runtime.h>

typedef __attribute__((ext_vector_type(16))) __bf16          v16bf;
typedef __attribute__((ext_vector_type(8)))  float           v8f;
typedef __attribute__((ext_vector_type(8)))  unsigned short  ushort8v;
typedef __attribute__((ext_vector_type(4)))  float           float4v;
typedef __attribute__((ext_vector_type(2)))  unsigned int    uint2v;

// Problem constants (from the reference)
constexpr int Td = 20;     // time bins (M per trial)
constexpr int Pd = 512;    // channels  (K)
constexpr int Nd = 1024;   // neurons   (N)

constexpr int KC    = 64;   // K per main-loop iteration (2 WMMA k-steps)
constexpr int NTILE = 128;  // N columns per workgroup (8 waves: 2 M-tiles x 4 N-slots x 2 slices)
constexpr int ASU   = Pd + 8;   // A row stride in ushorts (1040 B, 16B aligned, bank-skewed)
constexpr int ASD   = ASU / 2;  // ... in dwords
constexpr int BSU   = KC + 8;   // Bt row stride in ushorts (144 B, 16B aligned)
constexpr int BSD   = BSU / 2;  // ... in dwords

union FragU { v16bf v; ushort8v h[2]; };

__device__ inline float trunc_bf(float a) {
    return __uint_as_float(__float_as_uint(a) & 0xffff0000u);
}
// pack bf16(a) into [15:0], bf16(b) into [31:16]
__device__ inline unsigned pack_hi(float a, float b) {
    return (__float_as_uint(a) >> 16) | (__float_as_uint(b) & 0xffff0000u);
}
__device__ inline ushort8v ld8(const unsigned* arr, int uoff) {
    return *(const ushort8v*)((const unsigned short*)arr + uoff);
}

__device__ inline void stage_w_load(const float* We, int kc2, int tid,
                                    float4v w0[4], float4v w1[4]) {
    #pragma unroll
    for (int i = 0; i < 4; ++i) {                 // 64x128 f32 tile, 2 rows x 4 cols per chunk
        int c = tid + i * 256;
        int n = (c & 31) * 4;
        int k = (c >> 5) * 2;
        w0[i] = *(const float4v*)(We + (size_t)(kc2 + k) * Nd + n);
        w1[i] = *(const float4v*)(We + (size_t)(kc2 + k + 1) * Nd + n);
    }
}
__device__ inline void stage_w_store(int tid, const float4v w0[4], const float4v w1[4],
                                     unsigned* bhi, unsigned* blo) {
    #pragma unroll
    for (int i = 0; i < 4; ++i) {
        int c  = tid + i * 256;
        int n  = (c & 31) * 4;
        int kd = (c >> 5);                        // k/2: dword index within transposed row
        #pragma unroll
        for (int j = 0; j < 4; ++j) {
            float a = w0[i][j], b = w1[i][j];
            bhi[(n + j) * BSD + kd] = pack_hi(a, b);
            blo[(n + j) * BSD + kd] = pack_hi(a - trunc_bf(a), b - trunc_bf(b));
        }
    }
}

__global__ __launch_bounds__(256)
void stitch_decoder_wmma(const float* __restrict__ X,     // [B, T, P]
                         const float* __restrict__ W,     // [E, P, N]
                         const float* __restrict__ Bias,  // [E, N]
                         const int*   __restrict__ Eid,   // [B]
                         float*       __restrict__ Out)   // [B, T, N]
{
    __shared__ __align__(16) unsigned sAhi[Td * ASD];
    __shared__ __align__(16) unsigned sAlo[Td * ASD];
    __shared__ __align__(16) unsigned sBhi[2][NTILE * BSD];   // ping-pong
    __shared__ __align__(16) unsigned sBlo[2][NTILE * BSD];

    const int tid   = threadIdx.x;
    const int b     = blockIdx.x;
    const int ncol0 = blockIdx.y * NTILE;
    const int e     = Eid[b];

    const float* Xb = X + (size_t)b * Td * Pd;
    const float* We = W + (size_t)e * Pd * Nd + ncol0;

    // ---- Stage A: x[b] (20x512 f32) -> hi/lo bf16 (packed dword pairs) in LDS ----
    #pragma unroll
    for (int i = 0; i < (Td * Pd) / (4 * 256); ++i) {         // 10 float4 per thread
        int elem = (tid + i * 256) * 4;
        int r = elem >> 9;
        int k = elem & (Pd - 1);
        float4v xv = *(const float4v*)(Xb + r * Pd + k);
        uint2v h = { pack_hi(xv[0], xv[1]), pack_hi(xv[2], xv[3]) };
        uint2v l = { pack_hi(xv[0] - trunc_bf(xv[0]), xv[1] - trunc_bf(xv[1])),
                     pack_hi(xv[2] - trunc_bf(xv[2]), xv[3] - trunc_bf(xv[3])) };
        *(uint2v*)&sAhi[r * ASD + k / 2] = h;
        *(uint2v*)&sAlo[r * ASD + k / 2] = l;
    }
    // ---- Prologue: stage W tile 0 into buffer 0 ----
    {
        float4v w0[4], w1[4];
        stage_w_load(We, 0, tid, w0, w1);
        stage_w_store(tid, w0, w1, sBhi[0], sBlo[0]);
    }
    __syncthreads();

    const int wave = tid >> 5;
    const int lane = tid & 31;
    const int mi   = wave >> 2;      // 0..1 -> rows 0-15 / 16-31 (rows >= 20 masked)
    const int ni   = wave & 3;       // 0..3 -> 16-col slot
    const int half = lane >> 4;      // ISA half-wave selector
    const int l16  = lane & 15;
    const int arow = mi * 16 + l16;

    v8f acc0 = {}, acc1 = {};
    constexpr int nIter = Pd / KC;   // 8

    for (int it = 0; it < nIter; ++it) {
        const int  kc      = it * KC;
        const int  cur     = it & 1;
        const bool hasNext = (it + 1 < nIter);

        // distance-2 prefetch (256 cachelines of the 64x128 f32 tile -> 1 per thread)
        if (it + 2 < nIter)
            __builtin_prefetch(We + (size_t)(kc + 2 * KC + (tid >> 2)) * Nd + (tid & 3) * 32, 0, 1);

        // software pipeline: issue next tile's global loads before this tile's WMMAs
        float4v w0[4], w1[4];
        if (hasNext) stage_w_load(We, kc + KC, tid, w0, w1);

        #pragma unroll
        for (int ks = 0; ks < 2; ++ks) {
            // A fragments (contiguous k order, same as B: half 0 -> K 0-15; half 1 -> K 16-31)
            FragU ah, al;
            int aoff = arow * ASU + kc + ks * 32 + half * 16;
            if (arow < Td) {
                ah.h[0] = ld8(sAhi, aoff);      ah.h[1] = ld8(sAhi, aoff + 8);
                al.h[0] = ld8(sAlo, aoff);      al.h[1] = ld8(sAlo, aoff + 8);
            } else {
                ushort8v z = {};
                ah.h[0] = z; ah.h[1] = z; al.h[0] = z; al.h[1] = z;
            }
            #pragma unroll
            for (int s = 0; s < 2; ++s) {
                int bcol = s * 64 + ni * 16 + l16;
                int boff = bcol * BSU + ks * 32 + half * 16;
                FragU bh, bl;
                bh.h[0] = ld8(sBhi[cur], boff); bh.h[1] = ld8(sBhi[cur], boff + 8);
                bl.h[0] = ld8(sBlo[cur], boff); bl.h[1] = ld8(sBlo[cur], boff + 8);
                v8f& acc = s ? acc1 : acc0;
                // bf16x3 split-precision: al*bh + ah*bl + ah*bh
                acc = __builtin_amdgcn_wmma_f32_16x16x32_bf16(false, al.v, false, bh.v, (short)0, acc, false, false);
                acc = __builtin_amdgcn_wmma_f32_16x16x32_bf16(false, ah.v, false, bl.v, (short)0, acc, false, false);
                acc = __builtin_amdgcn_wmma_f32_16x16x32_bf16(false, ah.v, false, bh.v, (short)0, acc, false, false);
            }
        }

        if (hasNext) stage_w_store(tid, w0, w1, sBhi[cur ^ 1], sBlo[cur ^ 1]);
        __syncthreads();   // one barrier per iteration (ping-pong buffers)
    }

    // ---- Epilogue: bias + store (C layout: VGPR v -> M = 16*mi + 8*half + v, N = l16) ----
    float* outb = Out + (size_t)b * Td * Nd;
    #pragma unroll
    for (int s = 0; s < 2; ++s) {
        int col = ncol0 + s * 64 + ni * 16 + l16;
        float bias = Bias[(size_t)e * Nd + col];
        v8f acc = s ? acc1 : acc0;
        #pragma unroll
        for (int v = 0; v < 8; ++v) {
            int row = mi * 16 + half * 8 + v;
            if (row < Td) outb[row * Nd + col] = acc[v] + bias;
        }
    }
}

extern "C" void kernel_launch(void* const* d_in, const int* in_sizes, int n_in,
                              void* d_out, int out_size, void* d_ws, size_t ws_size,
                              hipStream_t stream) {
    (void)n_in; (void)out_size; (void)d_ws; (void)ws_size;
    const float* X    = (const float*)d_in[0];
    const float* W    = (const float*)d_in[1];
    const float* Bias = (const float*)d_in[2];
    const int*   Eid  = (const int*)d_in[3];
    float*       Out  = (float*)d_out;

    const int nTrials = in_sizes[3];           // B = 512
    dim3 grid(nTrials, Nd / NTILE);            // 512 x 8 workgroups
    stitch_decoder_wmma<<<grid, 256, 0, stream>>>(X, W, Bias, Eid, Out);
}
